// PermutationSinkhornLayer_24807731102178
// MI455X (gfx1250) — matrix operand order repaired and checked by reading, not compiled, not run on hardware
//
#include <hip/hip_runtime.h>

// ---------------------------------------------------------------------------
// PermutationSinkhornLayer for MI455X (gfx1250, wave32, WMMA)
// B=16, S=384, D=1024, HEADS=8, hd=128, AR=2, TEMP=1, mask all-ones.
// GEMMs run on v_wmma_f32_16x16x32_f16, 16x64 tiles/wave (4 accumulators
// sharing one A fragment), software-pipelined: next k-step's fragments are
// loaded before the current step's 4 WMMAs so global loads overlap the
// matrix pipe instead of s_wait_loadcnt 0x0 before every WMMA.
// Weights are converted f32->f16 TRANSPOSED so every WMMA operand is two
// contiguous global_load_b128. Sinkhorn (200 it) and Bethe (30 it) are
// L2-resident ping-pong passes (18.9 MB << 192 MB L2). exp() after Sinkhorn
// is skipped: monotonic, only feeds argmax matching.
// ---------------------------------------------------------------------------

#define S_LEN 384
#define SS    (S_LEN * S_LEN)   // 147456
#define BATCH 16
#define REPS  32                // BATCH * AR
#define D_INP 1024
#define H1N   512
#define H2N   2048
#define EPSF  1e-10f

typedef __attribute__((ext_vector_type(16))) _Float16 v16h;
typedef __attribute__((ext_vector_type(8)))  float    v8f;

__device__ __forceinline__ v8f wmma16(v16h a, v16h b, v8f c) {
  return __builtin_amdgcn_wmma_f32_16x16x32_f16(
      /*neg_a=*/false, a, /*neg_b=*/false, b,
      /*c_mod=*/(short)0, c, /*reuse_a=*/false, /*reuse_b=*/false);
}

// A-fragment: 16x32 f16, row-major source (M x K). lane&15 = M row; lane>>4
// selects K half (contiguous: lanes 0-15 K=0..15, lanes 16-31 K=16..31), matching B.
// Contiguous per-lane 32B span -> two global_load_b128.
__device__ __forceinline__ v16h load_a_frag(const _Float16* A, int lda,
                                            int m0, int k0, int lane) {
  int m  = m0 + (lane & 15);
  int kh = (lane >> 4) << 4;
  const _Float16* p = A + (size_t)m * lda + k0;
  v16h a;
#pragma unroll
  for (int v = 0; v < 4; ++v) { int K = kh + 2 * v;            a[2*v] = p[K]; a[2*v+1] = p[K+1]; }
#pragma unroll
  for (int v = 4; v < 8; ++v) { int K = kh + 2 * v; a[2*v] = p[K]; a[2*v+1] = p[K+1]; }
  return a;
}

// B-fragment for B = Bt^T where Bt is (N x K) row-major: B[k][n] = Bt[n][k].
// lane&15 = N col; lanes 0-15 K=0..15, lanes 16-31 K=16..31 (VGPR v->K=2v,2v+1).
// Contiguous per-lane 32B span -> two global_load_b128.
__device__ __forceinline__ v16h load_bt_frag(const _Float16* Bt, int ldk,
                                             int k0, int n0, int lane) {
  int n  = n0 + (lane & 15);
  int kh = (lane >> 4) << 4;
  const _Float16* p = Bt + (size_t)n * ldk + k0 + kh;
  v16h b;
#pragma unroll
  for (int v = 0; v < 8; ++v) { b[2*v] = p[2*v]; b[2*v+1] = p[2*v+1]; }
  return b;
}

// ---------------- LayerNorm -> f16 ----------------
__global__ void k_ln(const float* __restrict__ X, const float* __restrict__ g,
                     const float* __restrict__ bb, _Float16* __restrict__ Y) {
  __shared__ float red[256];
  int row = blockIdx.x;                       // 0..6143
  const float* x = X + (size_t)row * D_INP;
  int tid = threadIdx.x;
  float s = 0.f, s2 = 0.f;
  for (int i = tid; i < D_INP; i += 256) { float v = x[i]; s += v; s2 += v * v; }
  red[tid] = s; __syncthreads();
  for (int o = 128; o > 0; o >>= 1) { if (tid < o) red[tid] += red[tid + o]; __syncthreads(); }
  float mean = red[0] * (1.f / D_INP); __syncthreads();
  red[tid] = s2; __syncthreads();
  for (int o = 128; o > 0; o >>= 1) { if (tid < o) red[tid] += red[tid + o]; __syncthreads(); }
  float var  = red[0] * (1.f / D_INP) - mean * mean;
  float rinv = rsqrtf(var + 0.001f);
  for (int i = tid; i < D_INP; i += 256) {
    float v = (x[i] - mean) * rinv * g[i] + bb[i];
    Y[(size_t)row * D_INP + i] = (_Float16)v;
  }
}

// convert f32 (K x N) -> f16 transposed (N x K), 32x32 LDS tiles
__global__ void k_cvt_t(const float* __restrict__ X, _Float16* __restrict__ Y,
                        int K, int N) {
  __shared__ float tile[32][33];
  int k0 = blockIdx.x * 32, n0 = blockIdx.y * 32;
  int tx = threadIdx.x & 31, ty = threadIdx.x >> 5;   // 256 thr: 32 x 8
  for (int i = ty; i < 32; i += 8) tile[i][tx] = X[(size_t)(k0 + i) * N + n0 + tx];
  __syncthreads();
  for (int i = ty; i < 32; i += 8) Y[(size_t)(n0 + i) * K + k0 + tx] = (_Float16)tile[tx][i];
}

// ---------------- WMMA GEMM: C = act(A@Bt^T + bias) -> f16, 16x64 tiles -----
__device__ __forceinline__ void store_tile_f16(v8f acc, const float* __restrict__ bias,
                                               _Float16* __restrict__ C, int N,
                                               int tm, int tn, int lane, int relu) {
  int n  = tn + (lane & 15);
  int mh = (lane >> 4) << 3;
  float bv = bias[n];
#pragma unroll
  for (int r = 0; r < 8; ++r) {
    float x = acc[r] + bv;
    if (relu) x = fmaxf(x, 0.f);
    C[(size_t)(tm + r + mh) * N + n] = (_Float16)x;
  }
}

__global__ void k_gemm(const _Float16* __restrict__ A, const _Float16* __restrict__ Bt,
                       const float* __restrict__ bias, _Float16* __restrict__ C,
                       int M, int N, int K, int relu) {
  int wave = blockIdx.x * (blockDim.x >> 5) + (threadIdx.x >> 5);
  int lane = threadIdx.x & 31;
  int tilesN = N >> 6;                 // 64-wide tiles
  int tm = (wave / tilesN) << 4;
  int tn = (wave % tilesN) << 6;
  if (tm >= M) return;
  // prologue: fragments for k0 = 0
  v16h af = load_a_frag(A, K, tm, 0, lane);
  v16h b0 = load_bt_frag(Bt, K, 0, tn,      lane);
  v16h b1 = load_bt_frag(Bt, K, 0, tn + 16, lane);
  v16h b2 = load_bt_frag(Bt, K, 0, tn + 32, lane);
  v16h b3 = load_bt_frag(Bt, K, 0, tn + 48, lane);
  v8f a0 = {}, a1 = {}, a2 = {}, a3 = {};
  // pipelined: prefetch k0+32 before consuming k0
  for (int k0 = 0; k0 < K - 32; k0 += 32) {
    v16h afn = load_a_frag(A, K, tm, k0 + 32, lane);
    v16h c0  = load_bt_frag(Bt, K, k0 + 32, tn,      lane);
    v16h c1  = load_bt_frag(Bt, K, k0 + 32, tn + 16, lane);
    v16h c2  = load_bt_frag(Bt, K, k0 + 32, tn + 32, lane);
    v16h c3  = load_bt_frag(Bt, K, k0 + 32, tn + 48, lane);
    a0 = wmma16(af, b0, a0);
    a1 = wmma16(af, b1, a1);
    a2 = wmma16(af, b2, a2);
    a3 = wmma16(af, b3, a3);
    af = afn; b0 = c0; b1 = c1; b2 = c2; b3 = c3;
  }
  a0 = wmma16(af, b0, a0);
  a1 = wmma16(af, b1, a1);
  a2 = wmma16(af, b2, a2);
  a3 = wmma16(af, b3, a3);
  store_tile_f16(a0, bias, C, N, tm, tn,      lane, relu);
  store_tile_f16(a1, bias, C, N, tm, tn + 16, lane, relu);
  store_tile_f16(a2, bias, C, N, tm, tn + 32, lane, relu);
  store_tile_f16(a3, bias, C, N, tm, tn + 48, lane, relu);
}

// ---------------- scores: A[b] = (1/sqrt(hd)) * sum_h q_h @ k_h^T -----------
// H2 rows are (head-major) [q_h | k_h] blocks of 256. 32 pipelined steps:
// step st -> head st>>2, k-chunk (st&3)*32. K operand is naturally B^T.
__device__ __forceinline__ int q_col(int st) { return (st >> 2) * 256 + (st & 3) * 32; }
__device__ __forceinline__ int k_col(int st) { return q_col(st) + 128; }

__global__ void k_scores(const _Float16* __restrict__ H2, float* __restrict__ Abuf,
                         float* __restrict__ outA) {
  const int TM = S_LEN / 16;   // 24
  const int TN = S_LEN / 64;   // 6
  int wave = blockIdx.x * (blockDim.x >> 5) + (threadIdx.x >> 5);
  int lane = threadIdx.x & 31;
  int b    = wave / (TM * TN);
  int rem  = wave % (TM * TN);
  int tm   = (rem / TN) << 4;
  int tn   = (rem % TN) << 6;
  const _Float16* base = H2 + (size_t)b * S_LEN * H2N;
  v16h af = load_a_frag(base, H2N, tm, q_col(0), lane);
  v16h b0 = load_bt_frag(base, H2N, k_col(0), tn,      lane);
  v16h b1 = load_bt_frag(base, H2N, k_col(0), tn + 16, lane);
  v16h b2 = load_bt_frag(base, H2N, k_col(0), tn + 32, lane);
  v16h b3 = load_bt_frag(base, H2N, k_col(0), tn + 48, lane);
  v8f a0 = {}, a1 = {}, a2 = {}, a3 = {};
  for (int st = 0; st < 31; ++st) {
    v16h afn = load_a_frag(base, H2N, tm, q_col(st + 1), lane);
    v16h c0  = load_bt_frag(base, H2N, k_col(st + 1), tn,      lane);
    v16h c1  = load_bt_frag(base, H2N, k_col(st + 1), tn + 16, lane);
    v16h c2  = load_bt_frag(base, H2N, k_col(st + 1), tn + 32, lane);
    v16h c3  = load_bt_frag(base, H2N, k_col(st + 1), tn + 48, lane);
    a0 = wmma16(af, b0, a0);
    a1 = wmma16(af, b1, a1);
    a2 = wmma16(af, b2, a2);
    a3 = wmma16(af, b3, a3);
    af = afn; b0 = c0; b1 = c1; b2 = c2; b3 = c3;
  }
  a0 = wmma16(af, b0, a0);
  a1 = wmma16(af, b1, a1);
  a2 = wmma16(af, b2, a2);
  a3 = wmma16(af, b3, a3);
  const float scale = 0.08838834764831845f;   // 1/sqrt(128)
  int nl = lane & 15;
  int mh = (lane >> 4) << 3;
  v8f accs[4] = {a0, a1, a2, a3};
#pragma unroll
  for (int t = 0; t < 4; ++t) {
    int n = tn + t * 16 + nl;
#pragma unroll
    for (int r = 0; r < 8; ++r) {
      int m = tm + r + mh;
      float v = accs[t][r] * scale;
      Abuf[((size_t)b * S_LEN + m) * S_LEN + n] = v;
      // A_rep output (jnp.repeat along batch, AR=2)
      outA[((size_t)(2*b)     * S_LEN + m) * S_LEN + n] = v;
      outA[((size_t)(2*b + 1) * S_LEN + m) * S_LEN + n] = v;
    }
  }
}

// -------- per-batch stats: kl = -S^2 + sum(A) + sum(exp(-max(A,-20))), Amax --
__global__ void k_stats(const float* __restrict__ A, float* __restrict__ outKl,
                        float* __restrict__ Amax) {
  __shared__ float red[1024];
  int b = blockIdx.x, tid = threadIdx.x;
  const float* p = A + (size_t)b * SS;
  float s1 = 0.f, s2 = 0.f, mx = -3.4e38f;
  for (int i = tid; i < SS; i += 1024) {
    float v = p[i];
    s1 += v;
    s2 += __expf(-fmaxf(v, -20.f));
    mx  = fmaxf(mx, v);
  }
  red[tid] = s1; __syncthreads();
  for (int o = 512; o > 0; o >>= 1) { if (tid < o) red[tid] += red[tid + o]; __syncthreads(); }
  s1 = red[0]; __syncthreads();
  red[tid] = s2; __syncthreads();
  for (int o = 512; o > 0; o >>= 1) { if (tid < o) red[tid] += red[tid + o]; __syncthreads(); }
  s2 = red[0]; __syncthreads();
  red[tid] = mx; __syncthreads();
  for (int o = 512; o > 0; o >>= 1) { if (tid < o) red[tid] = fmaxf(red[tid], red[tid + o]); __syncthreads(); }
  if (tid == 0) {
    float kl = -(float)SS + s1 + s2;   // kl_t1=-S^2, lg=exp(lgamma(2))=1
    outKl[2*b] = kl; outKl[2*b + 1] = kl;
    Amax[b] = red[0];
  }
}

// ---------------- Bethe free energy ----------------
__global__ void k_bethe_init(const float* __restrict__ A, const float* __restrict__ Amax,
                             float* __restrict__ W, float* __restrict__ Bm) {
  __shared__ float red[128];
  int row = blockIdx.x;                 // 0..6143 (b*384+r)
  int b = row / S_LEN, tid = threadIdx.x;
  const float* a = A + (size_t)row * S_LEN;
  float mx = Amax[b], s = 0.f, w[3];
#pragma unroll
  for (int t = 0; t < 3; ++t) { float v = __expf(a[tid + 128*t] - mx); w[t] = v; s += v; }
  red[tid] = s; __syncthreads();
  for (int o = 64; o > 0; o >>= 1) { if (tid < o) red[tid] += red[tid + o]; __syncthreads(); }
  float inv = 1.f / (red[0] + EPSF);
#pragma unroll
  for (int t = 0; t < 3; ++t) {
    size_t i = (size_t)row * S_LEN + tid + 128*t;
    W[i] = w[t]; Bm[i] = w[t] * inv;
  }
}

__global__ void k_bethe_row(const float* __restrict__ W, const float* __restrict__ Bin,
                            float* __restrict__ Bout) {
  __shared__ float red[128];
  int row = blockIdx.x, tid = threadIdx.x;
  size_t base = (size_t)row * S_LEN;
  float y[3]; float s = 0.f;
#pragma unroll
  for (int t = 0; t < 3; ++t) {
    size_t i = base + tid + 128*t;
    float v = W[i] * fmaxf(1.f - Bin[i], 0.f);
    y[t] = v; s += v;
  }
  red[tid] = s; __syncthreads();
  for (int o = 64; o > 0; o >>= 1) { if (tid < o) red[tid] += red[tid + o]; __syncthreads(); }
  float inv = 1.f / (red[0] + EPSF);
#pragma unroll
  for (int t = 0; t < 3; ++t) Bout[base + tid + 128*t] = y[t] * inv;
}

__global__ void k_bethe_col(float* __restrict__ B) {
  // grid: BATCH * 12 blocks; 128 threads = 32 cols x 4 row-strides (coalesced)
  __shared__ float red[4][32];
  int blk = blockIdx.x, b = blk / 12, cg = blk % 12;
  int tx = threadIdx.x & 31, ty = threadIdx.x >> 5;
  float* base = B + (size_t)b * SS + cg * 32 + tx;
  float s = 0.f;
  for (int r = ty; r < S_LEN; r += 4) s += base[(size_t)r * S_LEN];
  red[ty][tx] = s; __syncthreads();
  if (ty == 0) red[0][tx] = 1.f / (red[0][tx] + red[1][tx] + red[2][tx] + red[3][tx] + EPSF);
  __syncthreads();
  float inv = red[0][tx];
  for (int r = ty; r < S_LEN; r += 4) base[(size_t)r * S_LEN] *= inv;
}

__global__ void k_bethe_final(const float* __restrict__ A, const float* __restrict__ Bm,
                              float* __restrict__ outNorm) {
  __shared__ float red[1024];
  int b = blockIdx.x, tid = threadIdx.x;
  const float* a = A  + (size_t)b * SS;
  const float* m = Bm + (size_t)b * SS;
  float t = 0.f;
  for (int i = tid; i < SS; i += 1024) {
    float bm = m[i];
    t += bm * a[i];                              // t1
    t -= bm * __logf(bm + EPSF);                 // t2
    float o = 1.f - bm;
    t += o * __logf(o + EPSF);                   // t3
  }
  red[tid] = t; __syncthreads();
  for (int o = 512; o > 0; o >>= 1) { if (tid < o) red[tid] += red[tid + o]; __syncthreads(); }
  if (tid == 0) { outNorm[2*b] = red[0]; outNorm[2*b + 1] = red[0]; }
}

// ---------------- Sinkhorn (log domain; final exp skipped — monotonic) ------
__global__ void k_sink_init(const float* __restrict__ A, const float* __restrict__ g,
                            float* __restrict__ Ps) {
  size_t i = (size_t)blockIdx.x * blockDim.x + threadIdx.x;
  if (i >= (size_t)REPS * SS) return;
  size_t r = i / SS, rem = i - r * SS, b = r >> 1;
  Ps[i] = A[b * SS + rem] + g[i];     // TEMP = 1
}

__global__ void k_sink_row(float* __restrict__ Ps) {
  __shared__ float red[128];
  float* x = Ps + (size_t)blockIdx.x * S_LEN;   // grid = REPS*S_LEN
  int tid = threadIdx.x;
  float v0 = x[tid], v1 = x[tid + 128], v2 = x[tid + 256];
  float mx = fmaxf(fmaxf(v0, v1), v2);
  red[tid] = mx; __syncthreads();
  for (int o = 64; o > 0; o >>= 1) { if (tid < o) red[tid] = fmaxf(red[tid], red[tid + o]); __syncthreads(); }
  mx = red[0]; __syncthreads();
  float s = __expf(v0 - mx) + __expf(v1 - mx) + __expf(v2 - mx);
  red[tid] = s; __syncthreads();
  for (int o = 64; o > 0; o >>= 1) { if (tid < o) red[tid] += red[tid + o]; __syncthreads(); }
  float lse = mx + __logf(red[0]);
  x[tid] = v0 - lse; x[tid + 128] = v1 - lse; x[tid + 256] = v2 - lse;
}

__global__ void k_sink_col(float* __restrict__ Ps) {
  // grid: REPS * 12 blocks; 128 threads = 32 cols x 4 row-strides (coalesced)
  __shared__ float red[4][32];
  int blk = blockIdx.x, r = blk / 12, cg = blk % 12;
  int tx = threadIdx.x & 31, ty = threadIdx.x >> 5;
  float* base = Ps + (size_t)r * SS + cg * 32 + tx;
  float mx = -3.4e38f;
  for (int row = ty; row < S_LEN; row += 4) mx = fmaxf(mx, base[(size_t)row * S_LEN]);
  red[ty][tx] = mx; __syncthreads();
  if (ty == 0) red[0][tx] = fmaxf(fmaxf(red[0][tx], red[1][tx]), fmaxf(red[2][tx], red[3][tx]));
  __syncthreads();
  mx = red[0][tx]; __syncthreads();
  float s = 0.f;
  for (int row = ty; row < S_LEN; row += 4) s += __expf(base[(size_t)row * S_LEN] - mx);
  red[ty][tx] = s; __syncthreads();
  if (ty == 0) red[0][tx] = mx + __logf(red[0][tx] + red[1][tx] + red[2][tx] + red[3][tx]);
  __syncthreads();
  float lse = red[0][tx];
  for (int row = ty; row < S_LEN; row += 4) base[(size_t)row * S_LEN] -= lse;
}

__global__ void k_zero(float* __restrict__ p, size_t n) {
  size_t i = (size_t)blockIdx.x * blockDim.x + threadIdx.x;
  if (i < n) p[i] = 0.f;
}

// ---------------- greedy matching (sequential argmax) + log_nom -------------
__global__ void k_match(const float* __restrict__ Ps, const float* __restrict__ Abuf,
                        float* __restrict__ P, float* __restrict__ outNom) {
  __shared__ float bv[1024];
  __shared__ int   bi[1024];
  __shared__ unsigned char rok[S_LEN], cok[S_LEN];
  __shared__ float acc;
  int r = blockIdx.x, b = r >> 1, tid = threadIdx.x;
  const float* X  = Ps   + (size_t)r * SS;
  const float* Ab = Abuf + (size_t)b * SS;
  float*       Pr = P    + (size_t)r * SS;
  for (int i = tid; i < S_LEN; i += 1024) { rok[i] = 1; cok[i] = 1; }
  if (tid == 0) acc = 0.f;
  __syncthreads();
  for (int step = 0; step < S_LEN; ++step) {
    float best = -1.0e30f; int bidx = 0;
    int rr = tid / S_LEN, cc = tid - rr * S_LEN;
    for (int i = tid; i < SS; i += 1024) {
      if (rok[rr] && cok[cc]) { float v = X[i]; if (v > best) { best = v; bidx = i; } }
      rr += 2; cc += 256;                         // += 1024 = 2*384 + 256
      if (cc >= S_LEN) { cc -= S_LEN; ++rr; }
    }
    bv[tid] = best; bi[tid] = bidx; __syncthreads();
    for (int o = 512; o > 0; o >>= 1) {
      if (tid < o && bv[tid + o] > bv[tid]) { bv[tid] = bv[tid + o]; bi[tid] = bi[tid + o]; }
      __syncthreads();
    }
    if (tid == 0) {
      int idx = bi[0]; int rw = idx / S_LEN, cl = idx - rw * S_LEN;
      Pr[idx] = 1.f; rok[rw] = 0; cok[cl] = 0; acc += Ab[idx];
    }
    __syncthreads();
  }
  if (tid == 0) outNom[r] = acc;
}

// ---------------------------------------------------------------------------
extern "C" void kernel_launch(void* const* d_in, const int* in_sizes, int n_in,
                              void* d_out, int out_size, void* d_ws, size_t ws_size,
                              hipStream_t stream) {
  const float* queries = (const float*)d_in[0];
  const float* ln_g    = (const float*)d_in[1];
  const float* ln_b    = (const float*)d_in[2];
  const float* w1      = (const float*)d_in[3];
  const float* b1      = (const float*)d_in[4];
  const float* w2      = (const float*)d_in[5];
  const float* b2      = (const float*)d_in[6];
  const float* gumbel  = (const float*)d_in[7];
  // d_in[8]=queries_mask (all ones), d_in[9]=pretrain_done(=1), d_in[10]=AR(=2)

  char* ws = (char*)d_ws;
  size_t off = 0;
  auto take = [&](size_t bytes) -> char* {
    off = (off + 255) & ~(size_t)255;
    char* p = ws + off; off += bytes; return p;
  };
  _Float16* xln  = (_Float16*)take((size_t)6144 * D_INP * 2);
  _Float16* w1t  = (_Float16*)take((size_t)D_INP * H1N * 2);   // H1N x D_INP
  _Float16* w2t  = (_Float16*)take((size_t)H1N * H2N * 2);     // H2N x H1N
  _Float16* h1h  = (_Float16*)take((size_t)6144 * H1N * 2);
  _Float16* h2h  = (_Float16*)take((size_t)6144 * H2N * 2);
  float* Abuf    = (float*)take((size_t)BATCH * SS * 4);
  float* Wb      = (float*)take((size_t)BATCH * SS * 4);
  float* BmA     = (float*)take((size_t)BATCH * SS * 4);
  float* BmB     = (float*)take((size_t)BATCH * SS * 4);
  float* Amax    = (float*)take(256);
  float* Ps      = (float*)take((size_t)REPS * SS * 4);

  float* outP    = (float*)d_out;
  float* outA    = outP + (size_t)REPS * SS;
  float* outKl   = outA + (size_t)REPS * SS;
  float* outNom  = outKl + REPS;
  float* outNorm = outNom + REPS;

  // 1) LayerNorm -> f16; weights f32 -> f16 transposed (N x K)
  k_ln<<<6144, 256, 0, stream>>>(queries, ln_g, ln_b, xln);
  k_cvt_t<<<dim3(D_INP / 32, H1N / 32), 256, 0, stream>>>(w1, w1t, D_INP, H1N);
  k_cvt_t<<<dim3(H1N / 32, H2N / 32), 256, 0, stream>>>(w2, w2t, H1N, H2N);

  // 2) MLP via WMMA (16x64 tiles/wave, pipelined): relu(x@w1+b1) -> h1; h1@w2+b2 -> h2
  k_gemm<<<(6144 / 16) * (H1N / 64) / 8, 256, 0, stream>>>(xln, w1t, b1, h1h, 6144, H1N, D_INP, 1);
  k_gemm<<<(6144 / 16) * (H2N / 64) / 8, 256, 0, stream>>>(h1h, w2t, b2, h2h, 6144, H2N, H1N, 0);

  // 3) scores A = sum_h q k^T / sqrt(hd) via WMMA; also writes A_rep output
  k_scores<<<(BATCH * 24 * 6) / 8, 256, 0, stream>>>(h2h, Abuf, outA);

  // 4) kl + per-batch max
  k_stats<<<BATCH, 1024, 0, stream>>>(Abuf, outKl, Amax);

  // 5) Bethe: init + 30 row/col iterations (ping-pong), then log_norm
  k_bethe_init<<<BATCH * S_LEN, 128, 0, stream>>>(Abuf, Amax, Wb, BmA);
  float* bcur = BmA; float* bnext = BmB;
  for (int it = 0; it < 30; ++it) {
    k_bethe_row<<<BATCH * S_LEN, 128, 0, stream>>>(Wb, bcur, bnext);
    k_bethe_col<<<BATCH * 12, 128, 0, stream>>>(bnext);
    float* t = bcur; bcur = bnext; bnext = t;
  }
  k_bethe_final<<<BATCH, 1024, 0, stream>>>(Abuf, bcur, outNorm);

  // 6) Sinkhorn: 200 x (row LSE, col LSE); stays L2-resident (18.9 MB)
  k_sink_init<<<(REPS * SS) / 256, 256, 0, stream>>>(Abuf, gumbel, Ps);
  for (int it = 0; it < 200; ++it) {
    k_sink_row<<<REPS * S_LEN, 128, 0, stream>>>(Ps);
    k_sink_col<<<REPS * 12, 128, 0, stream>>>(Ps);
  }
  // final exp skipped: monotonic, P_soft only feeds argmax matching

  // 7) greedy matching -> P (zero-init) and log_nom
  k_zero<<<(REPS * SS) / 256, 256, 0, stream>>>(outP, (size_t)REPS * SS);
  k_match<<<REPS, 1024, 0, stream>>>(Ps, Abuf, outP, outNom);

  (void)in_sizes; (void)n_in; (void)out_size; (void)ws_size;
}
